// SelfAttention_71081708748998
// MI455X (gfx1250) — matrix operand order repaired and checked by reading, not compiled, not run on hardware
//
#include <hip/hip_runtime.h>
#include <hip/hip_bf16.h>

// ---------------- types ----------------
typedef __attribute__((ext_vector_type(16))) __bf16 v16bf;
typedef __attribute__((ext_vector_type(8)))  __bf16 v8bf;
typedef __attribute__((ext_vector_type(8)))  float  v8f;
typedef __attribute__((ext_vector_type(4)))  int    v4i_t;

#define D_MODEL 1024
#define N_HEADS 16
#define D_HEAD  64
#define BATCH   4
#define SEQ     2048
#define NTOK    (BATCH*SEQ)          /* 8192 */

// ---- CDNA5 async global->LDS path (guarded; falls back to direct copies) ----
#if defined(__AMDGCN__) && __has_builtin(__builtin_amdgcn_global_load_async_to_lds_b128)
#define HAVE_ASYNC_LDS 1
#endif

__device__ __forceinline__ void async_copy16(const unsigned short* g, unsigned short* l) {
#ifdef HAVE_ASYNC_LDS
  __builtin_amdgcn_global_load_async_to_lds_b128(
      (__attribute__((address_space(1))) v4i_t*)(v4i_t*)(unsigned short*)g,
      (__attribute__((address_space(3))) v4i_t*)(v4i_t*)l,
      0, 0);
#else
  *(uint4*)l = *(const uint4*)g;
#endif
}

__device__ __forceinline__ void wait_async_copies() {
#ifdef HAVE_ASYNC_LDS
#if __has_builtin(__builtin_amdgcn_s_wait_asynccnt)
  __builtin_amdgcn_s_wait_asynccnt(0);
#else
  asm volatile("s_wait_asynccnt 0x0" ::: "memory");
#endif
#endif
}

__device__ __forceinline__ unsigned short f2bf(float f) {
  unsigned int u = __float_as_uint(f);
  u += 0x7fffu + ((u >> 16) & 1u);   // round-to-nearest-even
  return (unsigned short)(u >> 16);
}

__device__ __forceinline__ v16bf make_v16(v8bf lo, v8bf hi) {
  return __builtin_shufflevector(lo, hi, 0,1,2,3,4,5,6,7,8,9,10,11,12,13,14,15);
}

// ---------------- conversion kernels ----------------
__global__ void k_convert(const float* __restrict__ src,
                          unsigned short* __restrict__ dst, int n) {
  int i = blockIdx.x * blockDim.x + threadIdx.x;
  if (i < n) dst[i] = f2bf(src[i]);
}

// wt[n][k] = w[k][n]  (store W^T so B-fragments are contiguous in K)
__global__ void k_transpose_convert(const float* __restrict__ w,
                                    unsigned short* __restrict__ wt) {
  int i = blockIdx.x * blockDim.x + threadIdx.x;   // 0 .. 1024*1024-1
  int n = i >> 10, k = i & 1023;
  wt[i] = f2bf(w[(size_t)k * D_MODEL + n]);
}

// ---------------- shared GEMM main loop ----------------
// C[BM x BN] tile of A(bf16 MxK row-major) * W^T(bf16 NxK row-major)
// 256 threads = 8 waves in 2x4; each wave: 4x2 16x16 fragments (64x32).
#define BM 128
#define BN 128
#define BK 64
#define LDSS 72   /* padded row stride (ushorts): 144B, 16B-aligned, bank-friendly */

__device__ __forceinline__ void gemm_stage(
    const unsigned short* __restrict__ A,
    const unsigned short* __restrict__ Bt,
    int K, int m0, int n0, int kt,
    unsigned short* as, unsigned short* bs, int tid)
{
#pragma unroll
  for (int t = 0; t < 4; ++t) {
    int chunk = tid + t * 256;                 // 0..1023, coalesced
    int r = chunk >> 3, c8 = chunk & 7;
    async_copy16(A  + (size_t)(m0 + r) * K + kt * BK + c8 * 8, &as[r * LDSS + c8 * 8]);
    async_copy16(Bt + (size_t)(n0 + r) * K + kt * BK + c8 * 8, &bs[r * LDSS + c8 * 8]);
  }
}

__device__ __forceinline__ void gemm_mainloop(
    const unsigned short* __restrict__ A,
    const unsigned short* __restrict__ Bt,
    int K, int m0, int n0, v8f (&c)[4][2])
{
  __shared__ unsigned short As[2][BM * LDSS];
  __shared__ unsigned short Bs[2][BN * LDSS];

  const int tid  = threadIdx.x;
  const int wave = tid >> 5, lane = tid & 31;
  const int half = lane >> 4, l15 = lane & 15;
  const int wm = wave >> 2, wn = wave & 3;

  v8f zero = {};
#pragma unroll
  for (int fm = 0; fm < 4; ++fm)
#pragma unroll
    for (int fn = 0; fn < 2; ++fn) c[fm][fn] = zero;

  const int nkt = K / BK;
  gemm_stage(A, Bt, K, m0, n0, 0, As[0], Bs[0], tid);   // prologue: tile 0 -> buf 0

  for (int kt = 0; kt < nkt; ++kt) {
    const int cur = kt & 1;
    wait_async_copies();          // buf[cur] complete (this wave's copies)
    __syncthreads();              // all waves done copying + done reading buf[1-cur]
    if (kt + 1 < nkt)             // overlap: async-stage next tile during compute
      gemm_stage(A, Bt, K, m0, n0, kt + 1, As[1 - cur], Bs[1 - cur], tid);

    const unsigned short* as = As[cur];
    const unsigned short* bs = Bs[cur];
#pragma unroll
    for (int ks = 0; ks < 2; ++ks) {
      v16bf a[4], b[2];
#pragma unroll
      for (int fm = 0; fm < 4; ++fm) {
        int r = wm * 64 + fm * 16 + l15;
        v8bf lo = *(const v8bf*)&as[r * LDSS + ks * 32 + 16 * half];
        v8bf hi = *(const v8bf*)&as[r * LDSS + ks * 32 + 16 * half + 8];
        a[fm] = make_v16(lo, hi);
      }
#pragma unroll
      for (int fn = 0; fn < 2; ++fn) {
        int r = wn * 32 + fn * 16 + l15;
        v8bf lo = *(const v8bf*)&bs[r * LDSS + ks * 32 + 16 * half];
        v8bf hi = *(const v8bf*)&bs[r * LDSS + ks * 32 + 16 * half + 8];
        b[fn] = make_v16(lo, hi);
      }
#pragma unroll
      for (int fm = 0; fm < 4; ++fm)
#pragma unroll
        for (int fn = 0; fn < 2; ++fn)
          c[fm][fn] = __builtin_amdgcn_wmma_f32_16x16x32_bf16(
              false, a[fm], false, b[fn], (short)0, c[fm][fn], false, false);
    }
  }
}

// fused Q/K/V projection: blockIdx.z selects weight/bias/output
__global__ __launch_bounds__(256) void k_gemm_qkv(
    const unsigned short* __restrict__ xb,
    const unsigned short* __restrict__ wqT,
    const unsigned short* __restrict__ wkT,
    const unsigned short* __restrict__ wvT,
    const float* __restrict__ bq, const float* __restrict__ bk2,
    const float* __restrict__ bv,
    unsigned short* __restrict__ qb, unsigned short* __restrict__ kb2,
    unsigned short* __restrict__ vtb)
{
  const int z = blockIdx.z;
  const unsigned short* Bt = (z == 0) ? wqT : (z == 1) ? wkT : wvT;
  const float* bias        = (z == 0) ? bq  : (z == 1) ? bk2 : bv;
  unsigned short* outB     = (z == 0) ? qb  : (z == 1) ? kb2 : vtb;
  const int m0 = blockIdx.y * BM, n0 = blockIdx.x * BN;

  v8f c[4][2];
  gemm_mainloop(xb, Bt, D_MODEL, m0, n0, c);

  const int tid  = threadIdx.x;
  const int wave = tid >> 5, lane = tid & 31;
  const int half = lane >> 4, l15 = lane & 15;
  const int wm = wave >> 2, wn = wave & 3;
#pragma unroll
  for (int fn = 0; fn < 2; ++fn) {
    int n = n0 + wn * 32 + fn * 16 + l15;
    float bvv = bias[n];
    int h = n >> 6, d = n & 63;
#pragma unroll
    for (int fm = 0; fm < 4; ++fm) {
#pragma unroll
      for (int i = 0; i < 8; ++i) {
        int m = m0 + wm * 64 + fm * 16 + i + 8 * half;
        int bI = m >> 11, s = m & 2047;
        float v = c[fm][fn][i] + bvv;
        size_t idx = (z != 2)
            ? ((size_t)(bI * N_HEADS + h) * SEQ + s) * D_HEAD + d       // [b,h,s,d]
            : ((size_t)(bI * N_HEADS + h) * D_HEAD + d) * SEQ + s;      // [b,h,d,s]
        outB[idx] = f2bf(v);
      }
    }
  }
}

// final projection: fp32 row-major output
__global__ __launch_bounds__(256) void k_gemm_out(
    const unsigned short* __restrict__ A,
    const unsigned short* __restrict__ Bt,
    const float* __restrict__ bias,
    float* __restrict__ outF)
{
  const int m0 = blockIdx.y * BM, n0 = blockIdx.x * BN;
  v8f c[4][2];
  gemm_mainloop(A, Bt, D_MODEL, m0, n0, c);

  const int tid  = threadIdx.x;
  const int wave = tid >> 5, lane = tid & 31;
  const int half = lane >> 4, l15 = lane & 15;
  const int wm = wave >> 2, wn = wave & 3;
#pragma unroll
  for (int fn = 0; fn < 2; ++fn) {
    int n = n0 + wn * 32 + fn * 16 + l15;
    float bvv = bias[n];
#pragma unroll
    for (int fm = 0; fm < 4; ++fm) {
#pragma unroll
      for (int i = 0; i < 8; ++i) {
        int m = m0 + wm * 64 + fm * 16 + i + 8 * half;
        outF[(size_t)m * D_MODEL + n] = c[fm][fn][i] + bvv;
      }
    }
  }
}

// ---------------- fused flash attention ----------------
// grid (S/64, B*H), block 128 (4 waves). Each wave owns 16 query rows.
// Q,K: [b*h][S][64] bf16 ; Vt: [b*h][64][S] bf16 ; ctx out: [b][s][h][64] bf16
#define KVS 72   /* padded K/V tile row stride (ushorts) */

__device__ __forceinline__ void attn_stage(
    const unsigned short* __restrict__ kbase,
    const unsigned short* __restrict__ vbase,
    int kb, unsigned short* ks, unsigned short* vs, int tid)
{
#pragma unroll
  for (int t = 0; t < 4; ++t) {
    int chunk = tid + t * 128;                 // 0..511
    int r = chunk >> 3, c8 = chunk & 7;
    async_copy16(kbase + (size_t)(kb * 64 + r) * D_HEAD + c8 * 8, &ks[r * KVS + c8 * 8]);
    async_copy16(vbase + (size_t)r * SEQ + kb * 64 + c8 * 8,      &vs[r * KVS + c8 * 8]);
  }
}

__global__ __launch_bounds__(128) void k_attn(
    const unsigned short* __restrict__ Q,
    const unsigned short* __restrict__ Kk,
    const unsigned short* __restrict__ Vt,
    unsigned short* __restrict__ ctx)
{
  __shared__ unsigned short Ks[2][64 * KVS];   // [s_k][d], padded
  __shared__ unsigned short Vs[2][64 * KVS];   // [d][s_k], padded
  __shared__ unsigned short Ps[4 * 16 * 72];   // per-wave P staging, 144B rows

  const int tid  = threadIdx.x;
  const int wave = tid >> 5, lane = tid & 31;
  const int half = lane >> 4, l15 = lane & 15;
  const int bh = blockIdx.y;
  const int q0 = blockIdx.x * 64 + wave * 16;

  const unsigned short* qbase = Q  + (size_t)bh * SEQ * D_HEAD;
  const unsigned short* kbase = Kk + (size_t)bh * SEQ * D_HEAD;
  const unsigned short* vbase = Vt + (size_t)bh * D_HEAD * SEQ;
  unsigned short* pbase = Ps + wave * 16 * 72;

  // Q A-fragments held in registers (d = 2 k-steps of 32)
  v16bf qf[2];
#pragma unroll
  for (int ks = 0; ks < 2; ++ks) {
    const unsigned short* p = qbase + (size_t)(q0 + l15) * D_HEAD + ks * 32 + 16 * half;
    qf[ks] = make_v16(*(const v8bf*)p, *(const v8bf*)(p + 8));
  }

  float mI[8], lI[8];
  v8f zero = {};
  v8f o[4];
#pragma unroll
  for (int i = 0; i < 8; ++i) { mI[i] = -__builtin_inff(); lI[i] = 0.f; }
#pragma unroll
  for (int j = 0; j < 4; ++j) o[j] = zero;

  const float scale = 0.125f;   // 1/sqrt(64)
  const int nb = SEQ / 64;

  attn_stage(kbase, vbase, 0, Ks[0], Vs[0], tid);   // prologue

  for (int kb = 0; kb < nb; ++kb) {
    const int cur = kb & 1;
    wait_async_copies();
    __syncthreads();
    if (kb + 1 < nb)
      attn_stage(kbase, vbase, kb + 1, Ks[1 - cur], Vs[1 - cur], tid);

    const unsigned short* ksb = Ks[cur];
    const unsigned short* vsb = Vs[cur];

    // scores S = Q (16x64) * K^T (64x64): 4 N-frags x 2 k-steps
    v8f sc[4];
#pragma unroll
    for (int j = 0; j < 4; ++j) sc[j] = zero;
#pragma unroll
    for (int ks = 0; ks < 2; ++ks) {
#pragma unroll
      for (int j = 0; j < 4; ++j) {
        int sl = j * 16 + l15;
        v16bf bk = make_v16(*(const v8bf*)&ksb[sl * KVS + ks * 32 + 16 * half],
                            *(const v8bf*)&ksb[sl * KVS + ks * 32 + 16 * half + 8]);
        sc[j] = __builtin_amdgcn_wmma_f32_16x16x32_bf16(
            false, qf[ks], false, bk, (short)0, sc[j], false, false);
      }
    }

    // online softmax: per VGPR index i, row = i + 8*half; reduce over 16 lanes
    float mNew[8];
#pragma unroll
    for (int i = 0; i < 8; ++i) {
      float mx = fmaxf(fmaxf(sc[0][i], sc[1][i]), fmaxf(sc[2][i], sc[3][i]));
      mx *= scale;
#pragma unroll
      for (int msk = 1; msk < 16; msk <<= 1)
        mx = fmaxf(mx, __shfl_xor(mx, msk, 32));
      mNew[i] = fmaxf(mI[i], mx);
      float alpha = __expf(mI[i] - mNew[i]);
      mI[i] = mNew[i];
      lI[i] *= alpha;
      o[0][i] *= alpha; o[1][i] *= alpha; o[2][i] *= alpha; o[3][i] *= alpha;
    }
#pragma unroll
    for (int i = 0; i < 8; ++i) {
      float s = 0.f;
#pragma unroll
      for (int j = 0; j < 4; ++j) {
        float p = __expf(sc[j][i] * scale - mNew[i]);
        sc[j][i] = p;
        s += p;
      }
#pragma unroll
      for (int msk = 1; msk < 16; msk <<= 1) s += __shfl_xor(s, msk, 32);
      lI[i] += s;
    }

    // stage P (bf16) into LDS in A-fragment-compatible layout
#pragma unroll
    for (int j = 0; j < 4; ++j)
#pragma unroll
      for (int i = 0; i < 8; ++i)
        pbase[(i + 8 * half) * 72 + j * 16 + l15] = f2bf(sc[j][i]);
    __syncthreads();

    // O += P (16x64) * V (64x64): A from Ps, B from Vs ([d][s_k] rows)
#pragma unroll
    for (int ks = 0; ks < 2; ++ks) {
      const unsigned short* pr = pbase + l15 * 72 + ks * 32 + 16 * half;
      v16bf ap = make_v16(*(const v8bf*)pr, *(const v8bf*)(pr + 8));
#pragma unroll
      for (int j = 0; j < 4; ++j) {
        int dl = j * 16 + l15;
        v16bf bv = make_v16(*(const v8bf*)&vsb[dl * KVS + ks * 32 + 16 * half],
                            *(const v8bf*)&vsb[dl * KVS + ks * 32 + 16 * half + 8]);
        o[j] = __builtin_amdgcn_wmma_f32_16x16x32_bf16(
            false, ap, false, bv, (short)0, o[j], false, false);
      }
    }
  }

  // finalize: O /= l, store ctx[b][s][h][d] as bf16
  const int bI = bh >> 4, h = bh & 15;
#pragma unroll
  for (int i = 0; i < 8; ++i) {
    float inv = 1.f / lI[i];
    int s = q0 + i + 8 * half;
#pragma unroll
    for (int j = 0; j < 4; ++j) {
      int d = j * 16 + l15;
      ctx[((size_t)(bI * SEQ + s) * N_HEADS + h) * D_HEAD + d] = f2bf(o[j][i] * inv);
    }
  }
}

// ---------------- launch ----------------
extern "C" void kernel_launch(void* const* d_in, const int* in_sizes, int n_in,
                              void* d_out, int out_size, void* d_ws, size_t ws_size,
                              hipStream_t stream) {
  const float* x  = (const float*)d_in[0];
  const float* wq = (const float*)d_in[1];
  const float* bq = (const float*)d_in[2];
  const float* wk = (const float*)d_in[3];
  const float* bk = (const float*)d_in[4];
  const float* wv = (const float*)d_in[5];
  const float* bv = (const float*)d_in[6];
  const float* wo = (const float*)d_in[7];
  const float* bo = (const float*)d_in[8];
  float* out = (float*)d_out;

  unsigned short* ws = (unsigned short*)d_ws;
  size_t off = 0;
  unsigned short* xb  = ws + off; off += (size_t)NTOK * D_MODEL;
  unsigned short* wqT = ws + off; off += (size_t)D_MODEL * D_MODEL;
  unsigned short* wkT = ws + off; off += (size_t)D_MODEL * D_MODEL;
  unsigned short* wvT = ws + off; off += (size_t)D_MODEL * D_MODEL;
  unsigned short* woT = ws + off; off += (size_t)D_MODEL * D_MODEL;
  unsigned short* qb  = ws + off; off += (size_t)NTOK * D_MODEL;
  unsigned short* kb  = ws + off; off += (size_t)NTOK * D_MODEL;
  unsigned short* vtb = ws + off; off += (size_t)NTOK * D_MODEL;
  unsigned short* cxb = ws + off; off += (size_t)NTOK * D_MODEL;

  const int nx = NTOK * D_MODEL;
  k_convert<<<(nx + 255) / 256, 256, 0, stream>>>(x, xb, nx);
  const int nw = D_MODEL * D_MODEL;
  k_transpose_convert<<<(nw + 255) / 256, 256, 0, stream>>>(wq, wqT);
  k_transpose_convert<<<(nw + 255) / 256, 256, 0, stream>>>(wk, wkT);
  k_transpose_convert<<<(nw + 255) / 256, 256, 0, stream>>>(wv, wvT);
  k_transpose_convert<<<(nw + 255) / 256, 256, 0, stream>>>(wo, woT);

  dim3 gq(D_MODEL / BN, NTOK / BM, 3);
  k_gemm_qkv<<<gq, 256, 0, stream>>>(xb, wqT, wkT, wvT, bq, bk, bv, qb, kb, vtb);

  k_attn<<<dim3(SEQ / 64, BATCH * N_HEADS), 128, 0, stream>>>(qb, kb, vtb, cxb);

  dim3 go(D_MODEL / BN, NTOK / BM);
  k_gemm_out<<<go, 256, 0, stream>>>(cxb, woT, bo, out);
}
